// SoftAggLayer_18433999635177
// MI455X (gfx1250) — compile-verified, matrix operand order checked by reading, not run on hardware
//
#include <hip/hip_runtime.h>
#include <hip/hip_bf16.h>

typedef __attribute__((ext_vector_type(16))) _Float16 v16h;
typedef __attribute__((ext_vector_type(8)))  float    v8f;

#define TB 256

// ---------------- utility kernels ----------------

__global__ void zero_f(float* p, int n) {
    int i = blockIdx.x * blockDim.x + threadIdx.x;
    if (i < n) p[i] = 0.0f;
}

__global__ void zero_f4(float4* p, int n4) {
    int i = blockIdx.x * blockDim.x + threadIdx.x;
    if (i < n4) p[i] = make_float4(0.f, 0.f, 0.f, 0.f);
}

__global__ void edge_prep(const long long* __restrict__ ei,
                          int* __restrict__ src, int* __restrict__ dst, int E) {
    int e = blockIdx.x * blockDim.x + threadIdx.x;
    if (e < E) {
        src[e] = (int)ei[e];
        dst[e] = (int)ei[(long long)E + e];
    }
}

__global__ void deg_kernel(const float* __restrict__ ea, const int* __restrict__ dst,
                           float* __restrict__ deg, int E) {
    int e = blockIdx.x * blockDim.x + threadIdx.x;
    if (e < E) atomicAdd(&deg[dst[e]], fabsf(ea[e]));
}

__global__ void dinv_kernel(const float* __restrict__ deg, float* __restrict__ dinv, int N) {
    int i = blockIdx.x * blockDim.x + threadIdx.x;
    if (i < N) {
        float d = deg[i];
        dinv[i] = (d > 0.0f) ? rsqrtf(d) : 0.0f;
    }
}

__global__ void norm_kernel(const float* __restrict__ ea, const int* __restrict__ src,
                            const int* __restrict__ dst, const float* __restrict__ dinv,
                            float* __restrict__ nrm, int E) {
    int e = blockIdx.x * blockDim.x + threadIdx.x;
    if (e < E) nrm[e] = dinv[src[e]] * fabsf(ea[e]) * dinv[dst[e]];
}

__global__ void init_feat(const float* __restrict__ x, float* __restrict__ h, int N) {
    int i = blockIdx.x * blockDim.x + threadIdx.x;
    if (i < N) {
        float4* p = (float4*)(h + (size_t)i * 8);
        p[0] = make_float4(x[i], 0.f, 0.f, 0.f);
        p[1] = make_float4(0.f, 0.f, 0.f, 0.f);
    }
}

// ---------------- propagation (the bandwidth-bound part) ----------------

// 1-channel variant for layer 1 (features live in channel 0 only)
__global__ void prop1(const float* __restrict__ hin, float* __restrict__ hout,
                      const int* __restrict__ src, const int* __restrict__ dst,
                      const float* __restrict__ nrm, int E) {
    int e = blockIdx.x * blockDim.x + threadIdx.x;
    if (e >= E) return;
    int s = src[e], d = dst[e];
    atomicAdd(hout + (size_t)d * 8, nrm[e] * hin[(size_t)s * 8]);
}

// 8-channel variant: two b128 gathers + 8 f32 atomics
__global__ void prop8(const float* __restrict__ hin, float* __restrict__ hout,
                      const int* __restrict__ src, const int* __restrict__ dst,
                      const float* __restrict__ nrm, int E) {
    int e = blockIdx.x * blockDim.x + threadIdx.x;
    if (e >= E) return;
    int s = src[e], d = dst[e];
    float nm = nrm[e];
    const float4* pin = (const float4*)(hin + (size_t)s * 8);
    float4 u = pin[0];
    float4 v = pin[1];
    float* o = hout + (size_t)d * 8;
    atomicAdd(o + 0, nm * u.x);
    atomicAdd(o + 1, nm * u.y);
    atomicAdd(o + 2, nm * u.z);
    atomicAdd(o + 3, nm * u.w);
    atomicAdd(o + 4, nm * v.x);
    atomicAdd(o + 5, nm * v.y);
    atomicAdd(o + 6, nm * v.z);
    atomicAdd(o + 7, nm * v.w);
}

// ---------------- WMMA combine ----------------

// One-time per layer: pack W (4,f_in,f_out) into the 32x16 f16 B fragment in
// lane-major order (lane*16 halves, WMMA K-split layout), and pad bias to 16.
__global__ void pack_wmma_b(const float* __restrict__ W, const float* __restrict__ bias,
                            int f_in, int f_out,
                            _Float16* __restrict__ Bfrag, float* __restrict__ biasPad) {
    int lane = threadIdx.x;                 // 32 threads
    if (lane >= 32) return;
    if (lane < 16) biasPad[lane] = (lane < f_out) ? bias[lane] : 0.0f;
    int col   = lane & 15;
    int hopLo = (lane < 16) ? 0 : 1;        // K in [0,16)
    int hopHi = (lane < 16) ? 2 : 3;        // K in [16,32)
    for (int j = 0; j < 8; ++j) {
        float wlo = 0.0f, whi = 0.0f;
        if (col < f_out && j < f_in) {
            wlo = W[(hopLo * f_in + j) * f_out + col];
            whi = W[(hopHi * f_in + j) * f_out + col];
        }
        Bfrag[lane * 16 + j]     = (_Float16)wlo;
        Bfrag[lane * 16 + 8 + j] = (_Float16)whi;
    }
}

// out(16x8) = [H0|P1|P2|P3](16x32) @ Bfrag(32x16) + bias, ReLU.
// One v_wmma_f32_16x16x32_f16 per 16-node tile; 8 waves per block.
// A-row loads are index-clamped (D row m depends only on A row m, and rows
// from clamped duplicates are never stored), so loads are unconditional.
__global__ void combine_wmma(const float* __restrict__ H0, const float* __restrict__ P1,
                             const float* __restrict__ P2, const float* __restrict__ P3,
                             const _Float16* __restrict__ Bfrag,
                             const float* __restrict__ biasPad,
                             float* __restrict__ Hout, int N) {
    int lane = threadIdx.x & 31;
    int wave = threadIdx.x >> 5;
    int tile = blockIdx.x * 8 + wave;
    if (tile * 16 >= N) return;             // wave-uniform

    int row   = lane & 15;
    int node  = tile * 16 + row;
    int nodeC = (node < N) ? node : (N - 1);

    // A layout (ISA 7.12.2, 16-bit, 16x32): lanes 0-15 hold K=0-7 & 16-23,
    // lanes 16-31 hold K=8-15 & 24-31.  K = 8*hop + channel.
    const float* loA = (lane < 16) ? H0 : P1;
    const float* hiA = (lane < 16) ? P2 : P3;

    const float4* lo4 = (const float4*)(loA + (size_t)nodeC * 8);
    const float4* hi4 = (const float4*)(hiA + (size_t)nodeC * 8);
    float4 l0 = lo4[0], l1 = lo4[1];
    float4 h0 = hi4[0], h1 = hi4[1];

    v16h a;
    a[0]  = (_Float16)l0.x; a[1]  = (_Float16)l0.y;
    a[2]  = (_Float16)l0.z; a[3]  = (_Float16)l0.w;
    a[4]  = (_Float16)l1.x; a[5]  = (_Float16)l1.y;
    a[6]  = (_Float16)l1.z; a[7]  = (_Float16)l1.w;
    a[8]  = (_Float16)h0.x; a[9]  = (_Float16)h0.y;
    a[10] = (_Float16)h0.z; a[11] = (_Float16)h0.w;
    a[12] = (_Float16)h1.x; a[13] = (_Float16)h1.y;
    a[14] = (_Float16)h1.z; a[15] = (_Float16)h1.w;

    // B fragment: precomputed lane-major -> unconditional 32B load
    v16h b = *(const v16h*)(Bfrag + lane * 16);

    v8f c = {};
    v8f d = __builtin_amdgcn_wmma_f32_16x16x32_f16(false, a, false, b,
                                                   (short)0, c, false, false);

    // C/D layout: VGPR r -> row r + 8*(lane>=16), col = lane&15
    int col = lane & 15;
    if (col < 8) {
        float bv = biasPad[col];
        int mbase = tile * 16 + ((lane < 16) ? 0 : 8);
        float* outp = Hout + (size_t)mbase * 8 + col;
        if (tile * 16 + 16 <= N) {          // wave-uniform full tile
            #pragma unroll
            for (int r = 0; r < 8; ++r) {
                float v = d[r] + bv;
                outp[(size_t)r * 8] = v > 0.0f ? v : 0.0f;
            }
        } else {                            // tail: guard stores only
            #pragma unroll
            for (int r = 0; r < 8; ++r) {
                if (mbase + r < N) {
                    float v = d[r] + bv;
                    outp[(size_t)r * 8] = v > 0.0f ? v : 0.0f;
                }
            }
        }
    }
}

// ---------------- normalization / selection ----------------

__global__ void reduce_sums(const float* __restrict__ H, int N, float* __restrict__ scal) {
    __shared__ float s1[TB], s2[TB];
    int tid = threadIdx.x;
    float a = 0.0f, b = 0.0f;
    for (int i = blockIdx.x * blockDim.x + tid; i < N; i += gridDim.x * blockDim.x) {
        float h = H[(size_t)i * 8];
        a += h; b += h * h;
    }
    s1[tid] = a; s2[tid] = b;
    __syncthreads();
    for (int s = blockDim.x >> 1; s > 0; s >>= 1) {
        if (tid < s) { s1[tid] += s1[tid + s]; s2[tid] += s2[tid + s]; }
        __syncthreads();
    }
    if (tid == 0) {
        atomicAdd(&scal[0], s1[0]);
        atomicAdd(&scal[1], s2[0]);
    }
}

__device__ __forceinline__ unsigned key_encode(float f) {
    unsigned u = __float_as_uint(f);
    return (u & 0x80000000u) ? ~u : (u | 0x80000000u);
}

__global__ void standardize(const float* __restrict__ H, const float* __restrict__ scal,
                            float* __restrict__ hstd, unsigned* __restrict__ key, int N) {
    int i = blockIdx.x * blockDim.x + threadIdx.x;
    if (i >= N) return;
    float mu  = scal[0] / (float)N;
    float var = scal[1] / (float)N - mu * mu;
    float h = (H[(size_t)i * 8] - mu) * rsqrtf(var + 1e-5f);
    hstd[i] = h;
    key[i]  = key_encode(h);
}

// single-workgroup exact k-th-largest radix select (4 x 8-bit passes)
__global__ void select_kth(const unsigned* __restrict__ key, int N,
                           const int* __restrict__ kptr, float* __restrict__ scal) {
    __shared__ unsigned hist[256];
    __shared__ unsigned prefix_s;
    __shared__ int k_s;
    int tid = threadIdx.x;
    if (tid == 0) { prefix_s = 0u; k_s = kptr[0]; }
    __syncthreads();
    for (int pass = 0; pass < 4; ++pass) {
        int shift = 24 - pass * 8;
        for (int i = tid; i < 256; i += blockDim.x) hist[i] = 0u;
        __syncthreads();
        unsigned pref = prefix_s;
        unsigned mask = (pass == 0) ? 0u : (0xFFFFFFFFu << (shift + 8));
        for (int i = tid; i < N; i += blockDim.x) {
            unsigned kk = key[i];
            if ((kk & mask) == pref)
                atomicAdd(&hist[(kk >> shift) & 255u], 1u);
        }
        __syncthreads();
        if (tid == 0) {
            int rem = k_s;
            int dsel = 0;
            for (int dgt = 255; dgt >= 0; --dgt) {
                int c = (int)hist[dgt];
                if (rem <= c) { dsel = dgt; break; }
                rem -= c;
            }
            k_s = rem;
            prefix_s = pref | ((unsigned)dsel << shift);
        }
        __syncthreads();
    }
    if (tid == 0) {
        unsigned kkey = prefix_s;
        unsigned u = (kkey & 0x80000000u) ? (kkey & 0x7FFFFFFFu) : ~kkey;
        ((unsigned*)scal)[2] = kkey;
        scal[3] = fabsf(__uint_as_float(u));
    }
}

__global__ void finalize(const float* __restrict__ hstd, const unsigned* __restrict__ key,
                         const float* __restrict__ scal, float* __restrict__ out, int N) {
    int i = blockIdx.x * blockDim.x + threadIdx.x;
    if (i >= N) return;
    unsigned kth = ((const unsigned*)scal)[2];
    float kabs = scal[3];
    float h = hstd[i];
    float m = (key[i] >= kth) ? 1.0f : -1.0f;
    float z = h - kabs + 0.1f * m;
    float soft = 1.0f / (1.0f + expf(-z));
    float2* o = (float2*)out;
    o[i] = make_float2(h, soft);
}

// ---------------- launch ----------------

extern "C" void kernel_launch(void* const* d_in, const int* in_sizes, int n_in,
                              void* d_out, int out_size, void* d_ws, size_t ws_size,
                              hipStream_t stream) {
    const float*     x   = (const float*)d_in[0];
    const float*     ea  = (const float*)d_in[1];
    const float*     W1  = (const float*)d_in[2];
    const float*     b1  = (const float*)d_in[3];
    const float*     W2  = (const float*)d_in[4];
    const float*     b2  = (const float*)d_in[5];
    const float*     W3  = (const float*)d_in[6];
    const float*     b3  = (const float*)d_in[7];
    const long long* ei  = (const long long*)d_in[8];
    const int*       kpt = (const int*)d_in[9];

    const int N = in_sizes[0];
    const int E = in_sizes[1];

    char* ws = (char*)d_ws;
    size_t off = 0;
    auto alloc = [&](size_t bytes) -> void* {
        void* p = ws + off;
        off += (bytes + 255) & ~(size_t)255;
        return p;
    };
    float*     deg   = (float*)alloc((size_t)N * 4);
    float*     dinv  = (float*)alloc((size_t)N * 4);
    float*     nrm   = (float*)alloc((size_t)E * 4);
    int*       src   = (int*)alloc((size_t)E * 4);
    int*       dst   = (int*)alloc((size_t)E * 4);
    float*     fA    = (float*)alloc((size_t)N * 8 * 4);
    float*     fB    = (float*)alloc((size_t)N * 8 * 4);
    float*     P1    = (float*)alloc((size_t)N * 8 * 4);
    float*     P2    = (float*)alloc((size_t)N * 8 * 4);
    float*     P3    = (float*)alloc((size_t)N * 8 * 4);
    float*     hstd  = (float*)alloc((size_t)N * 4);
    unsigned*  key   = (unsigned*)alloc((size_t)N * 4);
    _Float16*  Bf[3];
    float*     bp[3];
    for (int l = 0; l < 3; ++l) {
        Bf[l] = (_Float16*)alloc(32 * 16 * 2);
        bp[l] = (float*)alloc(16 * 4);
    }
    float* scal = (float*)alloc(64);
    (void)ws_size; (void)n_in; (void)out_size;

    int gN = (N + TB - 1) / TB;
    int gE = (E + TB - 1) / TB;
    int n4 = (N * 8) / 4;
    int g4 = (n4 + TB - 1) / TB;
    int tiles = (N + 15) / 16;
    int gC = (tiles + 7) / 8;

    zero_f<<<gN, TB, 0, stream>>>(deg, N);
    edge_prep<<<gE, TB, 0, stream>>>(ei, src, dst, E);
    deg_kernel<<<gE, TB, 0, stream>>>(ea, dst, deg, E);
    dinv_kernel<<<gN, TB, 0, stream>>>(deg, dinv, N);
    norm_kernel<<<gE, TB, 0, stream>>>(ea, src, dst, dinv, nrm, E);
    init_feat<<<gN, TB, 0, stream>>>(x, fA, N);

    const float* Ws[3] = {W1, W2, W3};
    const float* bs[3] = {b1, b2, b3};
    const int fin[3]  = {1, 8, 8};
    const int fout[3] = {8, 8, 1};
    for (int l = 0; l < 3; ++l)
        pack_wmma_b<<<1, 32, 0, stream>>>(Ws[l], bs[l], fin[l], fout[l], Bf[l], bp[l]);

    float* cur = fA;
    float* nxt = fB;
    for (int l = 0; l < 3; ++l) {
        zero_f4<<<g4, TB, 0, stream>>>((float4*)P1, n4);
        zero_f4<<<g4, TB, 0, stream>>>((float4*)P2, n4);
        zero_f4<<<g4, TB, 0, stream>>>((float4*)P3, n4);
        if (fin[l] == 1) {
            prop1<<<gE, TB, 0, stream>>>(cur, P1, src, dst, nrm, E);
            prop1<<<gE, TB, 0, stream>>>(P1,  P2, src, dst, nrm, E);
            prop1<<<gE, TB, 0, stream>>>(P2,  P3, src, dst, nrm, E);
        } else {
            prop8<<<gE, TB, 0, stream>>>(cur, P1, src, dst, nrm, E);
            prop8<<<gE, TB, 0, stream>>>(P1,  P2, src, dst, nrm, E);
            prop8<<<gE, TB, 0, stream>>>(P2,  P3, src, dst, nrm, E);
        }
        combine_wmma<<<gC, TB, 0, stream>>>(cur, P1, P2, P3, Bf[l], bp[l], nxt, N);
        float* t = cur; cur = nxt; nxt = t;
    }

    zero_f<<<1, 64, 0, stream>>>(scal, 2);
    reduce_sums<<<256, TB, 0, stream>>>(cur, N, scal);
    standardize<<<gN, TB, 0, stream>>>(cur, scal, hstd, key, N);
    select_kth<<<1, 1024, 0, stream>>>(key, N, kpt, scal);
    finalize<<<gN, TB, 0, stream>>>(hstd, key, scal, (float*)d_out, N);
}
